// MoEBlock_6571299963109
// MI455X (gfx1250) — matrix operand order repaired and checked by reading, not compiled, not run on hardware
//
#include <hip/hip_runtime.h>
#include <math.h>

// ---- problem constants (from reference setup_inputs) ----
#define T_TOK 8192
#define D_DIM 1024
#define H_DIM 2048
#define E_NUM 8

typedef __attribute__((ext_vector_type(16))) __bf16 v16bf;
typedef __attribute__((ext_vector_type(8)))  __bf16 v8bf;
typedef __attribute__((ext_vector_type(4)))  __bf16 v4bf;
typedef __attribute__((ext_vector_type(8)))  float  v8f;

#define WMMA_BF16(A, B, C) \
    __builtin_amdgcn_wmma_f32_16x16x32_bf16(false, (A), false, (B), (short)0, (C), false, false)

// ---- workspace layout (bytes, 256-aligned regions) ----
#define WS_COUNTS   0           // E ints (routing counts, double as cursors)
#define WS_PSUM     256         // E floats (softmax prob sums for aux loss)
#define WS_TOK      512         // E*T ints   (per-expert token lists, capacity T)
#define WS_WT       262656      // E*T floats (per-expert routing weights)
#define WS_HBF      524800      // T*D bf16   (normalized activations)
#define WS_W1P      17302016    // E*D*H bf16 (W1 in WMMA-B fragment layout)
#define WS_W2P      50856448    // E*H*D bf16 (W2 in WMMA-B fragment layout)

__device__ __forceinline__ float gelu_f(float x) {
    // jax.nn.gelu default (tanh approximation)
    return 0.5f * x * (1.0f + tanhf(0.7978845608028654f * (x + 0.044715f * x * x * x)));
}

// Build a 16x32 bf16 A-fragment from row-major LDS.
// Lanes 0-15 hold row M=lane, K = 0..15;
//                 lanes 16-31 hold row M=lane-16, K = 16..31.
__device__ __forceinline__ v16bf load_a_frag(const __bf16* s, int rowStride, int kbase, int lane) {
    int row = lane & 15;
    int hi8 = (lane >> 4) << 4;  // 0 or 16
    const __bf16* p = s + row * rowStride + kbase + hi8;
    union { v8bf v; __bf16 e[8]; } lo, hv;
    lo.v = *(const v8bf*)p;          // ds_load_b128
    hv.v = *(const v8bf*)(p + 8);    // ds_load_b128
    union { v16bf v; __bf16 e[16]; } r;
#pragma unroll
    for (int j = 0; j < 8; ++j) { r.e[j] = lo.e[j]; r.e[8 + j] = hv.e[j]; }
    return r.v;
}

// ---------------------------------------------------------------------------
// Kernel 1: repack W1 (E,D,H) and W2 (E,H,D) fp32 -> bf16 WMMA B-fragment tiles.
// B-frag layout (32x16 K x N tile, 512 elems): lane n (0-15) holds K=0..15 of
// column N=n; lane 16+n holds K=16..31 of column N=n; each lane's 16 values
// stored contiguously (32B) so the GEMM hot loop does 2x global_load_b128.
// ---------------------------------------------------------------------------
__global__ void __launch_bounds__(256) moe_prep_weights(const float* __restrict__ W1,
                                                        const float* __restrict__ W2,
                                                        __bf16* __restrict__ W1p,
                                                        __bf16* __restrict__ W2p) {
    long gid = (long)blockIdx.x * 256 + threadIdx.x;
    int lane = (int)(gid & 31);
    long tt  = gid >> 5;
    __bf16 tmp[16];
    if (gid < 1048576L) {                         // W1: E * (H/16=128) * (D/32=32) tiles
        int kc = (int)(tt & 31);
        int nt = (int)((tt >> 5) & 127);
        int e  = (int)(tt >> 12);
        int col  = nt * 16 + (lane & 15);
        int krow = kc * 32 + ((lane >> 4) << 4);
        const float* src = W1 + ((long)(e * D_DIM + krow)) * H_DIM + col;
#pragma unroll
        for (int j = 0; j < 16; ++j) tmp[j] = (__bf16)src[(long)j * H_DIM];
        __bf16* dst = W1p + ((((long)e * 128 + nt) * 32 + kc) << 9) + (lane << 4);
        *(v8bf*)dst       = *(v8bf*)tmp;
        *(v8bf*)(dst + 8) = *(v8bf*)(tmp + 8);
    } else {                                      // W2: E * (D/16=64) * (H/32=64) tiles
        long g2 = gid - 1048576L;
        tt = g2 >> 5;
        int kc = (int)(tt & 63);
        int nt = (int)((tt >> 6) & 63);
        int e  = (int)(tt >> 12);
        int col  = nt * 16 + (lane & 15);
        int krow = kc * 32 + ((lane >> 4) << 4);
        const float* src = W2 + ((long)(e * H_DIM + krow)) * D_DIM + col;
#pragma unroll
        for (int j = 0; j < 16; ++j) tmp[j] = (__bf16)src[(long)j * D_DIM];
        __bf16* dst = W2p + ((((long)e * 64 + nt) * 64 + kc) << 9) + (lane << 4);
        *(v8bf*)dst       = *(v8bf*)tmp;
        *(v8bf*)(dst + 8) = *(v8bf*)(tmp + 8);
    }
}

// ---------------------------------------------------------------------------
// Kernel 2: per-token LayerNorm + router logits + elastic top-k routing.
// One block (256 threads = 8 wave32) per token. Wave-level shuffle reductions.
// ---------------------------------------------------------------------------
__global__ void __launch_bounds__(256) moe_ln_route(const float* __restrict__ x,
                                                    const float* __restrict__ gamma,
                                                    const float* __restrict__ beta,
                                                    const float* __restrict__ Wg,
                                                    const int* __restrict__ topk_p,
                                                    const int* __restrict__ nact_p,
                                                    float* __restrict__ out,
                                                    __bf16* __restrict__ hbf,
                                                    int* __restrict__ counts,
                                                    float* __restrict__ psum,
                                                    int* __restrict__ tok,
                                                    float* __restrict__ wts) {
    int t    = blockIdx.x;
    int tid  = threadIdx.x;
    int wave = tid >> 5, lane = tid & 31;
    __shared__ float wsum[8], wsq[8], wlog[8][E_NUM];

    const float4 x4 = ((const float4*)(x + (long)t * D_DIM))[tid];
    float ls = x4.x + x4.y + x4.z + x4.w;
    float lq = x4.x * x4.x + x4.y * x4.y + x4.z * x4.z + x4.w * x4.w;
#pragma unroll
    for (int o = 16; o; o >>= 1) {
        ls += __shfl_xor(ls, o, 32);
        lq += __shfl_xor(lq, o, 32);
    }
    if (lane == 0) { wsum[wave] = ls; wsq[wave] = lq; }
    __syncthreads();
    float tot = 0.f, totq = 0.f;
#pragma unroll
    for (int w = 0; w < 8; ++w) { tot += wsum[w]; totq += wsq[w]; }

    float mu  = tot * (1.0f / D_DIM);
    float var = totq * (1.0f / D_DIM) - mu * mu;
    float rs  = rsqrtf(var + 1e-5f);
    const float4 g4 = ((const float4*)gamma)[tid];
    const float4 b4 = ((const float4*)beta)[tid];
    float h[4];
    h[0] = (x4.x - mu) * rs * g4.x + b4.x;
    h[1] = (x4.y - mu) * rs * g4.y + b4.y;
    h[2] = (x4.z - mu) * rs * g4.z + b4.z;
    h[3] = (x4.w - mu) * rs * g4.w + b4.w;

    union { v4bf v; __bf16 e[4]; } hb;
#pragma unroll
    for (int c = 0; c < 4; ++c) hb.e[c] = (__bf16)h[c];
    ((v4bf*)(hbf + (long)t * D_DIM))[tid] = hb.v;       // h in bf16 for WMMA
    ((float4*)(out + (long)t * D_DIM))[tid] = x4;       // residual init

    // router logits: Wg is (D, E) row-major, 8 floats per row
    float p[E_NUM];
#pragma unroll
    for (int e = 0; e < E_NUM; ++e) p[e] = 0.f;
#pragma unroll
    for (int c = 0; c < 4; ++c) {
        const float* wr = Wg + (long)(tid * 4 + c) * E_NUM;
#pragma unroll
        for (int e = 0; e < E_NUM; ++e) p[e] += h[c] * wr[e];
    }
#pragma unroll
    for (int e = 0; e < E_NUM; ++e) {
#pragma unroll
        for (int o = 16; o; o >>= 1) p[e] += __shfl_xor(p[e], o, 32);
    }
    if (lane == 0) {
#pragma unroll
        for (int e = 0; e < E_NUM; ++e) wlog[wave][e] = p[e];
    }
    __syncthreads();

    if (tid == 0) {
        float lg[E_NUM];
        int nA = *nact_p; nA = nA < 1 ? 1 : (nA > E_NUM ? E_NUM : nA);
        int k  = *topk_p; k  = k  < 1 ? 1 : (k  > E_NUM ? E_NUM : k);
#pragma unroll
        for (int e = 0; e < E_NUM; ++e) {
            float s = 0.f;
#pragma unroll
            for (int w = 0; w < 8; ++w) s += wlog[w][e];
            lg[e] = (e < nA) ? s : -3.0e38f;
        }
        // full softmax over masked logits (for aux loss pmean)
        float mx = lg[0];
#pragma unroll
        for (int e = 1; e < E_NUM; ++e) mx = fmaxf(mx, lg[e]);
        float pr[E_NUM], se = 0.f;
#pragma unroll
        for (int e = 0; e < E_NUM; ++e) { pr[e] = __expf(lg[e] - mx); se += pr[e]; }
        float inv = 1.0f / se;
#pragma unroll
        for (int e = 0; e < E_NUM; ++e)
            if (e < nA) atomicAdd(&psum[e], pr[e] * inv);
        // top-k selection + renormalized softmax among selected
        unsigned used = 0;
        int   ids[E_NUM];
        float vals[E_NUM];
        for (int i = 0; i < k; ++i) {
            int best = 0; float bv = -3.4e38f;
#pragma unroll
            for (int e = 0; e < E_NUM; ++e)
                if (!((used >> e) & 1u) && lg[e] > bv) { bv = lg[e]; best = e; }
            used |= 1u << best; ids[i] = best; vals[i] = bv;
        }
        float m2 = vals[0], s2 = 0.f;
        for (int i = 0; i < k; ++i) { vals[i] = __expf(vals[i] - m2); s2 += vals[i]; }
        float i2 = 1.0f / s2;
        for (int i = 0; i < k; ++i) {
            int e = ids[i];
            int pos = atomicAdd(&counts[e], 1);
            tok[e * T_TOK + pos] = t;
            wts[e * T_TOK + pos] = vals[i] * i2;
        }
    }
}

// ---------------------------------------------------------------------------
// Kernel 3: aux loss = E * sum_e( count[e]/T * psum[e]/T )
// ---------------------------------------------------------------------------
__global__ void moe_aux(const int* __restrict__ counts, const float* __restrict__ psum,
                        float* __restrict__ out) {
    __shared__ float s;
    if (threadIdx.x == 0) s = 0.f;
    __syncthreads();
    if (threadIdx.x < E_NUM) {
        float frac = (float)counts[threadIdx.x] * (1.0f / T_TOK);
        float pm   = psum[threadIdx.x] * (1.0f / T_TOK);
        atomicAdd(&s, frac * pm);
    }
    __syncthreads();
    if (threadIdx.x == 0) out[(long)T_TOK * D_DIM] = (float)E_NUM * s;
}

// ---------------------------------------------------------------------------
// Kernel 4: fused expert MLP for one (expert, 32-token tile), software-
// pipelined: next iteration's A (LDS) and B (L2-resident packed weights)
// fragments are fetched into separate locals before the current WMMAs issue,
// so >=2 B-fetches stay in flight (avoids one L2 latency per K-step).
// ---------------------------------------------------------------------------
__global__ void __launch_bounds__(256) moe_gemm(const __bf16* __restrict__ hbf,
                                                const __bf16* __restrict__ W1p,
                                                const __bf16* __restrict__ W2p,
                                                const float* __restrict__ b1,
                                                const float* __restrict__ b2,
                                                const int* __restrict__ counts,
                                                const int* __restrict__ tok,
                                                const float* __restrict__ wts,
                                                float* __restrict__ out) {
    int e  = blockIdx.x >> 8;        // expert
    int mt = blockIdx.x & 255;       // 32-token tile within expert list
    int nslot = counts[e];
    if (mt * 32 >= nslot) return;    // block-uniform exit: EXEC stays all-ones

    __shared__ __bf16 sh[32 * D_DIM];    // h tile, row-major      (64 KB)
    __shared__ __bf16 sh1[32 * 256];     // gelu(h@W1) chunk       (16 KB)
    __shared__ int    stok[32];
    __shared__ float  swt[32];

    int tid = threadIdx.x;
    if (tid < 32) {
        int slot = mt * 32 + tid;
        bool v = slot < nslot;
        stok[tid] = v ? tok[e * T_TOK + slot] : -1;
        swt[tid]  = v ? wts[e * T_TOK + slot] : 0.f;
    }
    __syncthreads();

    {   // stage h tile: thread -> (row = tid/8, 128 cols = 16 x b128)
        int row = tid >> 3;
        int cb  = (tid & 7) * 128;
        v8bf* dv = (v8bf*)(sh + row * D_DIM + cb);
        int tk = stok[row];
        if (tk >= 0) {
            const v8bf* sv = (const v8bf*)(hbf + (long)tk * D_DIM + cb);
#pragma unroll
            for (int c = 0; c < 16; ++c) dv[c] = sv[c];
        } else {
            union { v8bf v; __bf16 e2[8]; } z;
#pragma unroll
            for (int j = 0; j < 8; ++j) z.e2[j] = (__bf16)0.f;
#pragma unroll
            for (int c = 0; c < 16; ++c) dv[c] = z.v;
        }
    }
    __syncthreads();

    int wave = tid >> 5, lane = tid & 31;
    int nlane = lane & 15, rowOff = (lane >> 4) << 3;

    v8f acc[2][8];                       // [M-tile][N-tile] f32 accumulators
#pragma unroll
    for (int m = 0; m < 2; ++m)
#pragma unroll
        for (int j = 0; j < 8; ++j) acc[m][j] = (v8f){};

    for (int hc = 0; hc < H_DIM / 256; ++hc) {
        // ---- GEMM1: this wave's 32 hidden cols for both M-tiles, K = 1024 ----
        v8f c00 = {}, c01 = {}, c10 = {}, c11 = {};
        const __bf16* pb1 = W1p + ((((long)e * 128 + hc * 16 + wave * 2) * 32) << 9)
                                + (lane << 4);
        // prologue: fragments for kc = 0
        v16bf a0 = load_a_frag(sh,              D_DIM, 0, lane);
        v16bf a1 = load_a_frag(sh + 16 * D_DIM, D_DIM, 0, lane);
        v16bf b0 = *(const v16bf*)pb1;
        v16bf b1f = *(const v16bf*)(pb1 + (32 << 9));
#pragma unroll
        for (int kc = 0; kc < 32; ++kc) {
            v16bf na0 = a0, na1 = a1, nb0 = b0, nb1 = b1f;
            if (kc < 31) {               // prefetch kc+1 into fresh registers
                const __bf16* pn = pb1 + ((long)(kc + 1) << 9);
                na0 = load_a_frag(sh,              D_DIM, (kc + 1) * 32, lane);
                na1 = load_a_frag(sh + 16 * D_DIM, D_DIM, (kc + 1) * 32, lane);
                nb0 = *(const v16bf*)pn;
                nb1 = *(const v16bf*)(pn + (32 << 9));
            }
            c00 = WMMA_BF16(a0, b0,  c00);
            c01 = WMMA_BF16(a0, b1f, c01);
            c10 = WMMA_BF16(a1, b0,  c10);
            c11 = WMMA_BF16(a1, b1f, c11);
            a0 = na0; a1 = na1; b0 = nb0; b1f = nb1;
        }
        __syncthreads();   // all waves done reading sh1 from previous chunk
        {   // bias + GELU -> sh1 (bf16), both M-tiles
            int col0 = wave * 32 + nlane, col1 = col0 + 16;
            float bb0 = b1[e * H_DIM + hc * 256 + col0];
            float bb1 = b1[e * H_DIM + hc * 256 + col1];
            union { v8f v; float f[8]; } U00, U01, U10, U11;
            U00.v = c00; U01.v = c01; U10.v = c10; U11.v = c11;
#pragma unroll
            for (int r = 0; r < 8; ++r) {
                int row = r + rowOff;
                sh1[row * 256 + col0]        = (__bf16)gelu_f(U00.f[r] + bb0);
                sh1[row * 256 + col1]        = (__bf16)gelu_f(U01.f[r] + bb1);
                sh1[(row + 16) * 256 + col0] = (__bf16)gelu_f(U10.f[r] + bb0);
                sh1[(row + 16) * 256 + col1] = (__bf16)gelu_f(U11.f[r] + bb1);
            }
        }
        __syncthreads();
        // ---- GEMM2: rank-256 update of this wave's 128 output cols ----
        const __bf16* pb2 = W2p + ((((long)e * 64 + wave * 8) * 64 + hc * 8) << 9)
                                + (lane << 4);
        for (int i = 0; i < 8; ++i) {
            v16bf a20 = load_a_frag(sh1,            256, i * 32, lane);
            v16bf a21 = load_a_frag(sh1 + 16 * 256, 256, i * 32, lane);
            const __bf16* p = pb2 + ((long)i << 9);      // kc2 = hc*8 + i
            v16bf bcur = *(const v16bf*)p;               // j stride: 64 tiles
#pragma unroll
            for (int j = 0; j < 8; ++j) {
                v16bf bnxt = bcur;
                if (j < 7)                               // prefetch next N-tile
                    bnxt = *(const v16bf*)(p + ((long)(j + 1) * 64 << 9));
                acc[0][j] = WMMA_BF16(a20, bcur, acc[0][j]);   // B reused for
                acc[1][j] = WMMA_BF16(a21, bcur, acc[1][j]);   // both M-tiles
                bcur = bnxt;
            }
        }
    }

    // ---- epilogue: weighted scatter-add into residual ----
#pragma unroll
    for (int m = 0; m < 2; ++m)
#pragma unroll
    for (int j = 0; j < 8; ++j) {
        int col = wave * 128 + j * 16 + nlane;
        float bb = b2[e * D_DIM + col];
        union { v8f v; float f[8]; } U; U.v = acc[m][j];
#pragma unroll
        for (int r = 0; r < 8; ++r) {
            int row = m * 16 + r + rowOff;
            int tk = stok[row];
            if (tk >= 0)
                atomicAdd(out + (long)tk * D_DIM + col, swt[row] * (U.f[r] + bb));
        }
    }
}

// ---------------------------------------------------------------------------
extern "C" void kernel_launch(void* const* d_in, const int* in_sizes, int n_in,
                              void* d_out, int out_size, void* d_ws, size_t ws_size,
                              hipStream_t stream) {
    const float* x     = (const float*)d_in[0];
    const float* gamma = (const float*)d_in[1];
    const float* beta  = (const float*)d_in[2];
    const float* Wg    = (const float*)d_in[3];
    const float* W1    = (const float*)d_in[4];
    const float* b1    = (const float*)d_in[5];
    const float* W2    = (const float*)d_in[6];
    const float* b2    = (const float*)d_in[7];
    const int* topk    = (const int*)d_in[8];
    const int* nact    = (const int*)d_in[9];
    float* out = (float*)d_out;

    char* ws = (char*)d_ws;
    int*    counts = (int*)(ws + WS_COUNTS);
    float*  psum   = (float*)(ws + WS_PSUM);
    int*    tokL   = (int*)(ws + WS_TOK);
    float*  wtsL   = (float*)(ws + WS_WT);
    __bf16* hbf    = (__bf16*)(ws + WS_HBF);
    __bf16* W1p    = (__bf16*)(ws + WS_W1P);
    __bf16* W2p    = (__bf16*)(ws + WS_W2P);

    hipMemsetAsync(ws, 0, 512, stream);  // zero routing counters / psum each call
    moe_prep_weights<<<8192, 256, 0, stream>>>(W1, W2, W1p, W2p);
    moe_ln_route<<<T_TOK, 256, 0, stream>>>(x, gamma, beta, Wg, topk, nact,
                                            out, hbf, counts, psum, tokL, wtsL);
    moe_aux<<<1, 64, 0, stream>>>(counts, psum, out);
    moe_gemm<<<E_NUM * (T_TOK / 32), 256, 0, stream>>>(hbf, W1p, W2p, b1, b2,
                                                       counts, tokL, wtsL, out);
}
